// GlobalAttentionPooling_33294586478973
// MI455X (gfx1250) — matrix operand order repaired and checked by reading, not compiled, not run on hardware
//
#include <hip/hip_runtime.h>

#define D_IN   256
#define D_OUT  128
#define NSEG   1024

typedef __attribute__((ext_vector_type(16))) __bf16 v16bf;
typedef __attribute__((ext_vector_type(8)))  float  v8f;

// ---------- bf16 helpers (round-to-nearest-even split) ----------
static __device__ __forceinline__ __bf16 f2bf(float f) {
    unsigned u = __builtin_bit_cast(unsigned, f);
    unsigned r = u + 0x7FFFu + ((u >> 16) & 1u);
    unsigned short h = (unsigned short)(r >> 16);
    return __builtin_bit_cast(__bf16, h);
}
static __device__ __forceinline__ float bf2f(__bf16 b) {
    unsigned short h = __builtin_bit_cast(unsigned short, b);
    unsigned u = ((unsigned)h) << 16;
    return __builtin_bit_cast(float, u);
}

// ---------- kernel 1: kq = Wk @ q ; cb = bk . q ----------
__global__ void __launch_bounds__(256) k_prep(const float* __restrict__ Wk,
                                              const float* __restrict__ q,
                                              const float* __restrict__ bk,
                                              float* __restrict__ kq,
                                              float* __restrict__ cb) {
    __shared__ float qs[D_IN];
    __shared__ float red[256];
    int t = threadIdx.x;
    qs[t] = q[t];
    __syncthreads();
    const float* row = Wk + (size_t)t * D_IN;
    float acc = 0.f;
#pragma unroll 4
    for (int c = 0; c < D_IN; c += 4) {
        float4 wv = *(const float4*)(row + c);
        acc += wv.x * qs[c] + wv.y * qs[c + 1] + wv.z * qs[c + 2] + wv.w * qs[c + 3];
    }
    kq[t] = acc;
    red[t] = bk[t] * qs[t];
    __syncthreads();
    for (int ofs = 128; ofs > 0; ofs >>= 1) {
        if (t < ofs) red[t] += red[t + ofs];
        __syncthreads();
    }
    if (t == 0) cb[0] = red[0];
}

// ---------- kernel 2: per-segment online softmax + weighted row sum ----------
// One block (8 wave32) per segment. batch is sorted -> binary search range.
__global__ void __launch_bounds__(256) k_seg(const float* __restrict__ x,
                                             const int*   __restrict__ batch,
                                             const float* __restrict__ kq,
                                             const float* __restrict__ cbp,
                                             float* __restrict__ y,
                                             float* __restrict__ has,
                                             int N) {
    __shared__ int sr0, sr1;
    __shared__ float wM[8], wL[8];
    __shared__ float red[8][D_IN];

    const int s = blockIdx.x;
    const int t = threadIdx.x;
    const int lane = t & 31;
    const int w = t >> 5;

    if (t == 0) {
        int lo = 0, hi = N;
        while (lo < hi) { int mid = (lo + hi) >> 1; if (batch[mid] < s) lo = mid + 1; else hi = mid; }
        sr0 = lo;
        hi = N;
        while (lo < hi) { int mid = (lo + hi) >> 1; if (batch[mid] < s + 1) lo = mid + 1; else hi = mid; }
        sr1 = lo;
    }
    __syncthreads();
    const int r0 = sr0, r1 = sr1;
    const float cb = cbp[0];

    // lane owns columns [8*lane, 8*lane+8)
    const float4 kq0 = *(const float4*)(kq + lane * 8);
    const float4 kq1 = *(const float4*)(kq + lane * 8 + 4);

    // ---- pass 1: online softmax (per-wave running max m / sum l) ----
    float m = -__builtin_inff(), l = 0.f;
    for (int r = r0 + w; r < r1; r += 8) {
        const float* xr = x + (size_t)r * D_IN + lane * 8;
        float4 a = *(const float4*)(xr);
        float4 b = *(const float4*)(xr + 4);
        float p = a.x * kq0.x + a.y * kq0.y + a.z * kq0.z + a.w * kq0.w
                + b.x * kq1.x + b.y * kq1.y + b.z * kq1.z + b.w * kq1.w;
        for (int ofs = 16; ofs > 0; ofs >>= 1) p += __shfl_xor(p, ofs, 32);
        p += cb;
        float mn = fmaxf(m, p);
        l = l * __expf(m - mn) + __expf(p - mn);
        m = mn;
    }
    if (lane == 0) { wM[w] = m; wL[w] = l; }
    __syncthreads();

    // merge the 8 per-wave (m,l) pairs (redundant in every thread)
    float M = -__builtin_inff();
#pragma unroll
    for (int i = 0; i < 8; i++) M = fmaxf(M, wM[i]);
    float L = 0.f;
#pragma unroll
    for (int i = 0; i < 8; i++) L += wL[i] * __expf(wM[i] - M);
    const float invL = (L > 0.f) ? 1.0f / L : 0.f;

    // ---- pass 2: recompute score (rows are L2-hot), accumulate w_i * x_i ----
    float acc0 = 0.f, acc1 = 0.f, acc2 = 0.f, acc3 = 0.f;
    float acc4 = 0.f, acc5 = 0.f, acc6 = 0.f, acc7 = 0.f;
    for (int r = r0 + w; r < r1; r += 8) {
        const float* xr = x + (size_t)r * D_IN + lane * 8;
        float4 a = *(const float4*)(xr);
        float4 b = *(const float4*)(xr + 4);
        float p = a.x * kq0.x + a.y * kq0.y + a.z * kq0.z + a.w * kq0.w
                + b.x * kq1.x + b.y * kq1.y + b.z * kq1.z + b.w * kq1.w;
        for (int ofs = 16; ofs > 0; ofs >>= 1) p += __shfl_xor(p, ofs, 32);
        p += cb;
        float wgt = __expf(p - M) * invL;
        acc0 += wgt * a.x; acc1 += wgt * a.y; acc2 += wgt * a.z; acc3 += wgt * a.w;
        acc4 += wgt * b.x; acc5 += wgt * b.y; acc6 += wgt * b.z; acc7 += wgt * b.w;
    }
    float* rw = &red[w][lane * 8];
    rw[0] = acc0; rw[1] = acc1; rw[2] = acc2; rw[3] = acc3;
    rw[4] = acc4; rw[5] = acc5; rw[6] = acc6; rw[7] = acc7;
    __syncthreads();

    float sum = 0.f;
#pragma unroll
    for (int i = 0; i < 8; i++) sum += red[i][t];
    y[(size_t)s * D_IN + t] = sum;
    if (t == 0) has[s] = (r1 > r0) ? 1.f : 0.f;
}

// ---------- kernel 3: out = y @ Wv + has*bv via split-bf16 WMMA ----------
// y: [1024,256], Wv: [256,128] row-major, out: [1024,128]
// grid (64, 8) tiles of 16x16, one wave32 per tile, K=256 in steps of 32.
__global__ void __launch_bounds__(32) k_gemm(const float* __restrict__ y,
                                             const float* __restrict__ Wv,
                                             const float* __restrict__ bv,
                                             const float* __restrict__ has,
                                             float* __restrict__ out) {
    const int lane = threadIdx.x;       // 0..31
    const int half = lane >> 4;         // 0/1
    const int lm   = lane & 15;
    const int tm   = blockIdx.x * 16;   // M tile base (segment rows)
    const int tn   = blockIdx.y * 16;   // N tile base (output cols)

    v8f c = {};
    for (int k0 = 0; k0 < D_IN; k0 += 32) {
        // A fragment (16-bit A 16x32 layout): lane -> row tm+lm,
        // element e: K = k0 + (e/8)*16 + half*8 + (e%8)
        const float* arow = y + (size_t)(tm + lm) * D_IN + k0 + half * 8;
        float af[16];
#pragma unroll
        for (int i = 0; i < 8; i++) { af[i] = arow[i]; af[8 + i] = arow[16 + i]; }

        // B fragment (16-bit B 32x16 layout): lane -> col tn+lm,
        // element e: K = k0 + (e/8)*16 + half*8 + (e%8)
        float bf[16];
#pragma unroll
        for (int e = 0; e < 16; e++)
            bf[e] = Wv[(size_t)(k0 + (e / 8) * 16 + half * 8 + (e % 8)) * D_OUT + tn + lm];

        // split-bf16 (hi + residual) for ~fp32 accuracy with 3 products
        v16bf ahi, alo, bhi, blo;
#pragma unroll
        for (int e = 0; e < 16; e++) {
            __bf16 ah = f2bf(af[e]); ahi[e] = ah; alo[e] = f2bf(af[e] - bf2f(ah));
            __bf16 bh = f2bf(bf[e]); bhi[e] = bh; blo[e] = f2bf(bf[e] - bf2f(bh));
        }
        c = __builtin_amdgcn_wmma_f32_16x16x32_bf16(false, ahi, false, bhi, (short)0, c, false, false);
        c = __builtin_amdgcn_wmma_f32_16x16x32_bf16(false, ahi, false, blo, (short)0, c, false, false);
        c = __builtin_amdgcn_wmma_f32_16x16x32_bf16(false, alo, false, bhi, (short)0, c, false, false);
    }

    // C/D layout: VGPR r -> M = tm + half*8 + r, N = tn + lm
    const float bvn = bv[tn + lm];
#pragma unroll
    for (int r = 0; r < 8; r++) {
        int mrow = tm + half * 8 + r;
        out[(size_t)mrow * D_OUT + tn + lm] = c[r] + bvn * has[mrow];
    }
}

extern "C" void kernel_launch(void* const* d_in, const int* in_sizes, int n_in,
                              void* d_out, int out_size, void* d_ws, size_t ws_size,
                              hipStream_t stream) {
    const float* x     = (const float*)d_in[0];
    const int*   batch = (const int*)d_in[1];
    const float* query = (const float*)d_in[2];
    const float* Wk    = (const float*)d_in[3];
    const float* bk    = (const float*)d_in[4];
    const float* Wv    = (const float*)d_in[5];
    const float* bv    = (const float*)d_in[6];
    const int N = in_sizes[0] / D_IN;

    float* ws  = (float*)d_ws;
    float* y   = ws;                         // 1024*256 floats
    float* kq  = y + (size_t)NSEG * D_IN;    // 256 floats
    float* cb  = kq + D_IN;                  // 1 float (padded)
    float* has = cb + 16;                    // 1024 floats

    hipLaunchKernelGGL(k_prep, dim3(1), dim3(256), 0, stream, Wk, query, bk, kq, cb);
    hipLaunchKernelGGL(k_seg, dim3(NSEG), dim3(256), 0, stream, x, batch, kq, cb, y, has, N);
    hipLaunchKernelGGL(k_gemm, dim3(NSEG / 16, D_OUT / 16), dim3(32), 0, stream, y, Wv, bv, has, (float*)d_out);
}
